// RAFT_67783173865944
// MI455X (gfx1250) — compile-verified, matrix operand order checked by reading, not run on hardware
//
#include <hip/hip_runtime.h>

// ============================================================================
// RAFT forward (32 iters) for MI455X / gfx1250.
// All conv / correlation compute routed through v_wmma_f32_16x16x32_f16
// (f16 inputs, f32 accumulate), 2x2 register-blocked (16 FLOP/byte of
// fragment traffic). Corr pyramid (~87MB) sized to live in the 192MB L2.
// ============================================================================

typedef _Float16 half_t;
typedef __attribute__((ext_vector_type(16))) _Float16 v16h;
typedef __attribute__((ext_vector_type(8)))  float    v8f;

// ---- fixed geometry (B=1, 256x448 input, 1.5x zoom, stride-8 features) ----
constexpr int IHH = 256, IWW = 448;
constexpr int HZ  = 384, WZ  = 672;
constexpr int H1  = 192, W1  = 336;
constexpr int H2  = 96,  W2  = 168;
constexpr int FH  = 48,  FW  = 84;
constexpr long FHW = (long)FH * FW;          // 4032
constexpr int L1H = 24, L1W = 42;
constexpr int L2H = 12, L2W = 21;
constexpr int L3H = 6,  L3W = 10;
constexpr int RAFT_ITERS = 32;

#define GS_LOOP(i, n) \
  for (long i = (long)blockIdx.x * blockDim.x + threadIdx.x; i < (n); \
       i += (long)gridDim.x * blockDim.x)

// K-index mapping for the 16-bit A/B WMMA fragment (16x32), per ISA 7.12.2:
// lanes 0-15: VGPR0..3 -> K 0..7, VGPR4..7 -> K 16..23; lanes 16-31: +8.
__device__ __forceinline__ int wmma_kmap(int lane, int e) {
  return ((e < 8) ? e : (e + 8)) + ((lane >> 4) << 3);
}

// ---------------------------------------------------------------------------
// Pack kernels: produce A/B operands in the exact per-lane v16h layout so the
// GEMM inner loop is pure contiguous 32B loads + v_wmma.
// layout: pack[((mt*Kt + kt)*32 + lane)*16 + e]
// ---------------------------------------------------------------------------
__global__ void k_pack_a_im2col(const float* __restrict__ src, half_t* __restrict__ dst,
                                int Mt, int Kt, int M, int K, int OW,
                                int KH, int KW, int IHs, int IWs,
                                int stride, int padH, int padW) {
  long total = (long)Mt * Kt * 512;
  GS_LOOP(idx, total) {
    int e = (int)(idx & 15);
    int lane = (int)((idx >> 4) & 31);
    long t = idx >> 9;
    int kt = (int)(t % Kt);
    int mt = (int)(t / Kt);
    int m = mt * 16 + (lane & 15);
    int k = kt * 32 + wmma_kmap(lane, e);
    float v = 0.f;
    if (m < M && k < K) {
      int KHW = KH * KW;
      int ic = k / KHW;
      int r  = k - ic * KHW;
      int kh = r / KW, kw = r - (r / KW) * KW;
      int oh = m / OW, ow = m - oh * OW;
      int ih = oh * stride - padH + kh;
      int iw = ow * stride - padW + kw;
      if (ih >= 0 && ih < IHs && iw >= 0 && iw < IWs)
        v = src[((long)ic * IHs + ih) * IWs + iw];
    }
    dst[idx] = (half_t)v;
  }
}

// A[m][k] = src[k*ld + m] * alpha  (channel-major matrix as M-side operand)
__global__ void k_pack_a_mat(const float* __restrict__ src, half_t* __restrict__ dst,
                             int Mt, int Kt, int M, int K, long ld, float alpha) {
  long total = (long)Mt * Kt * 512;
  GS_LOOP(idx, total) {
    int e = (int)(idx & 15);
    int lane = (int)((idx >> 4) & 31);
    long t = idx >> 9;
    int kt = (int)(t % Kt);
    int mt = (int)(t / Kt);
    int m = mt * 16 + (lane & 15);
    int k = kt * 32 + wmma_kmap(lane, e);
    float v = (m < M && k < K) ? src[(long)k * ld + m] * alpha : 0.f;
    dst[idx] = (half_t)v;
  }
}

// B[k][n] = src[k*ks + n*ns] * alpha   (weights: ks=1, ns=K; features: ks=M, ns=1)
__global__ void k_pack_b(const float* __restrict__ src, half_t* __restrict__ dst,
                         int Nt, int Kt, int N, int K, long ks, long ns, float alpha) {
  long total = (long)Nt * Kt * 512;
  GS_LOOP(idx, total) {
    int e = (int)(idx & 15);
    int lane = (int)((idx >> 4) & 31);
    long t = idx >> 9;
    int kt = (int)(t % Kt);
    int nt = (int)(t / Kt);
    int n = nt * 16 + (lane & 15);
    int k = kt * 32 + wmma_kmap(lane, e);
    float v = (n < N && k < K) ? src[(long)k * ks + (long)n * ns] * alpha : 0.f;
    dst[idx] = (half_t)v;
  }
}

// ---------------------------------------------------------------------------
// WMMA GEMM, 2x2 register-blocked:
//   out[n*M + m] = act((sum_k A[m][k]B[k][n] + bias[n]) * alpha)
// Each wave owns a 32x32 output block (4 WMMA tiles); per K-step it loads
// 4 fragments (128B) and issues 4 v_wmma (64 KFLOP) -> 16 FLOP/B.
// Tile indices are wave-uniform => EXEC all-1s at every WMMA (ISA req).
// Edge tiles are clamped (duplicate tile => identical stores, deterministic).
// ACT: 0=none 1=relu 2=sigmoid 3=tanh
// ---------------------------------------------------------------------------
template <int ACT>
__device__ __forceinline__ void gemm_store_tile(const v8f& c, float* __restrict__ out,
                                                const float* __restrict__ bias,
                                                int M, int N, int mt, int nt,
                                                int lane, float alpha) {
  int n  = nt * 16 + (lane & 15);
  int m0 = mt * 16 + ((lane >> 4) << 3);
  if (n < N) {
    float bv = bias ? bias[n] : 0.f;
#pragma unroll
    for (int r = 0; r < 8; ++r) {
      int m = m0 + r;
      if (m < M) {
        float v = (c[r] + bv) * alpha;
        if (ACT == 1) v = fmaxf(v, 0.f);
        else if (ACT == 2) v = 1.f / (1.f + __expf(-v));
        else if (ACT == 3) v = tanhf(v);
        out[(long)n * M + m] = v;
      }
    }
  }
}

template <int ACT>
__global__ __launch_bounds__(256) void k_gemm(const half_t* __restrict__ A,
                                              const half_t* __restrict__ B,
                                              const float* __restrict__ bias,
                                              float* __restrict__ out,
                                              int M, int N, int Mt, int Nt, int Kt,
                                              float alpha) {
  int wpb  = blockDim.x >> 5;
  int wave = threadIdx.x >> 5;
  int lane = threadIdx.x & 31;
  int Mt2 = (Mt + 1) >> 1;
  int Nt2 = (Nt + 1) >> 1;
  long tiles = (long)Mt2 * Nt2;
  for (long tile = (long)blockIdx.x * wpb + wave; tile < tiles;
       tile += (long)gridDim.x * wpb) {
    int mt0 = ((int)(tile % Mt2)) * 2;           // M-major: waves in a block
    int nt0 = ((int)(tile / Mt2)) * 2;           // share the same B pair
    int mt1 = (mt0 + 1 < Mt) ? mt0 + 1 : mt0;
    int nt1 = (nt0 + 1 < Nt) ? nt0 + 1 : nt0;
    const half_t* ap0 = A + ((long)mt0 * Kt * 32 + lane) * 16;
    const half_t* ap1 = A + ((long)mt1 * Kt * 32 + lane) * 16;
    const half_t* bp0 = B + ((long)nt0 * Kt * 32 + lane) * 16;
    const half_t* bp1 = B + ((long)nt1 * Kt * 32 + lane) * 16;
    v8f c00 = {}, c01 = {}, c10 = {}, c11 = {};
    for (int kt = 0; kt < Kt; ++kt) {
      v16h a0 = *(const v16h*)ap0;
      v16h a1 = *(const v16h*)ap1;
      v16h b0 = *(const v16h*)bp0;
      v16h b1 = *(const v16h*)bp1;
      ap0 += 512; ap1 += 512; bp0 += 512; bp1 += 512;
      __builtin_prefetch(ap0, 0, 3);   // global_prefetch of next K-slab
      __builtin_prefetch(ap1, 0, 3);
      __builtin_prefetch(bp0, 0, 3);
      __builtin_prefetch(bp1, 0, 3);
      c00 = __builtin_amdgcn_wmma_f32_16x16x32_f16(false, a0, false, b0,
                                                   (short)0, c00, false, false);
      c01 = __builtin_amdgcn_wmma_f32_16x16x32_f16(false, a0, false, b1,
                                                   (short)0, c01, false, false);
      c10 = __builtin_amdgcn_wmma_f32_16x16x32_f16(false, a1, false, b0,
                                                   (short)0, c10, false, false);
      c11 = __builtin_amdgcn_wmma_f32_16x16x32_f16(false, a1, false, b1,
                                                   (short)0, c11, false, false);
    }
    gemm_store_tile<ACT>(c00, out, bias, M, N, mt0, nt0, lane, alpha);
    gemm_store_tile<ACT>(c01, out, bias, M, N, mt0, nt1, lane, alpha);
    gemm_store_tile<ACT>(c10, out, bias, M, N, mt1, nt0, lane, alpha);
    gemm_store_tile<ACT>(c11, out, bias, M, N, mt1, nt1, lane, alpha);
  }
}

// ---------------------------------------------------------------------------
// Elementwise / sampling kernels
// ---------------------------------------------------------------------------
__global__ void k_act_tanh(const float* s, float* d, long n) {
  GS_LOOP(i, n) d[i] = tanhf(s[i]);
}
__global__ void k_act_relu(const float* s, float* d, long n) {
  GS_LOOP(i, n) d[i] = fmaxf(s[i], 0.f);
}
__global__ void k_mul(const float* a, const float* b, float* d, long n) {
  GS_LOOP(i, n) d[i] = a[i] * b[i];
}
__global__ void k_gru_combine(float* h, const float* z, const float* q, long n) {
  GS_LOOP(i, n) h[i] = (1.f - z[i]) * h[i] + z[i] * q[i];
}
__global__ void k_add_inplace(float* a, const float* b, long n) {
  GS_LOOP(i, n) a[i] += b[i];
}
__global__ void k_coords_init(float* c) {
  GS_LOOP(i, 2 * FHW) {
    long ch = i / FHW, m = i - ch * FHW;
    c[i] = (ch == 0) ? (float)(m % FW) : (float)(m / FW);
  }
}
__global__ void k_flow_from_coords(const float* c1, float* f) {
  GS_LOOP(i, 2 * FHW) {
    long ch = i / FHW, m = i - ch * FHW;
    float base = (ch == 0) ? (float)(m % FW) : (float)(m / FW);
    f[i] = c1[i] - base;
  }
}

__global__ void k_resize_bilinear(const float* __restrict__ src, float* __restrict__ dst,
                                  int C, int SH, int SW, int DH, int DW,
                                  float scale, float bias) {
  long total = (long)C * DH * DW;
  GS_LOOP(idx, total) {
    int c = (int)(idx / ((long)DH * DW));
    long r = idx - (long)c * DH * DW;
    int oy = (int)(r / DW), ox = (int)(r - (long)(r / DW) * DW);
    float fy = (oy + 0.5f) * (float)SH / (float)DH - 0.5f;
    float fx = (ox + 0.5f) * (float)SW / (float)DW - 0.5f;
    float y0 = floorf(fy), x0 = floorf(fx);
    float wy = fy - y0, wx = fx - x0;
    int iy0 = (int)y0, ix0 = (int)x0;
    int iy1 = iy0 + 1, ix1 = ix0 + 1;
    iy0 = iy0 < 0 ? 0 : (iy0 > SH - 1 ? SH - 1 : iy0);
    iy1 = iy1 < 0 ? 0 : (iy1 > SH - 1 ? SH - 1 : iy1);
    ix0 = ix0 < 0 ? 0 : (ix0 > SW - 1 ? SW - 1 : ix0);
    ix1 = ix1 < 0 ? 0 : (ix1 > SW - 1 ? SW - 1 : ix1);
    const float* p = src + (long)c * SH * SW;
    float v = (1.f - wy) * ((1.f - wx) * p[(long)iy0 * SW + ix0] + wx * p[(long)iy0 * SW + ix1]) +
              wy * ((1.f - wx) * p[(long)iy1 * SW + ix0] + wx * p[(long)iy1 * SW + ix1]);
    dst[idx] = v * scale + bias;
  }
}

// count_include_pad average pool over C independent planes
__global__ void k_avgpool(const float* __restrict__ src, float* __restrict__ dst,
                          int C, int Hs, int Ws, int OH, int OW,
                          int k, int stride, int pad) {
  long total = (long)C * OH * OW;
  float inv = 1.f / (float)(k * k);
  GS_LOOP(idx, total) {
    int c = (int)(idx / ((long)OH * OW));
    long r = idx - (long)c * OH * OW;
    int oh = (int)(r / OW), ow = (int)(r - (long)(r / OW) * OW);
    const float* p = src + (long)c * Hs * Ws;
    float s = 0.f;
    for (int kh = 0; kh < k; ++kh) {
      int ih = oh * stride - pad + kh;
      if (ih < 0 || ih >= Hs) continue;
      for (int kw = 0; kw < k; ++kw) {
        int iw = ow * stride - pad + kw;
        if (iw < 0 || iw >= Ws) continue;
        s += p[(long)ih * Ws + iw];
      }
    }
    dst[idx] = s * inv;
  }
}

// RAFT correlation lookup: out[(lvl*81+j)*HW + m], bilinear, zero padding
__global__ void k_corr_lookup(const float* __restrict__ p0, const float* __restrict__ p1,
                              const float* __restrict__ p2, const float* __restrict__ p3,
                              const float* __restrict__ coords, float* __restrict__ out) {
  long total = 324 * FHW;
  GS_LOOP(idx, total) {
    long m = idx % FHW;
    int c = (int)(idx / FHW);
    int lvl = c / 81, j = c - lvl * 81;
    float dy = (float)(j / 9) - 4.f;
    float dx = (float)(j % 9) - 4.f;
    float x = coords[m], y = coords[FHW + m];
    float inv = 1.f / (float)(1 << lvl);
    float xl = x * inv + dx, yl = y * inv + dy;
    const float* img;
    int Hl, Wl;
    if (lvl == 0)      { img = p0; Hl = FH;  Wl = FW;  }
    else if (lvl == 1) { img = p1; Hl = L1H; Wl = L1W; }
    else if (lvl == 2) { img = p2; Hl = L2H; Wl = L2W; }
    else               { img = p3; Hl = L3H; Wl = L3W; }
    img += m * (long)Hl * Wl;
    float x0 = floorf(xl), y0 = floorf(yl);
    int ix = (int)x0, iy = (int)y0;
    float fx = xl - x0, fy = yl - y0;
    float v = 0.f;
#define TAP(xi, yi, w)                                                \
    do { int X = (xi), Y = (yi);                                      \
         if (X >= 0 && X <= Wl - 1 && Y >= 0 && Y <= Hl - 1)          \
           v += (w) * img[(long)Y * Wl + X]; } while (0)
    TAP(ix,     iy,     (1.f - fx) * (1.f - fy));
    TAP(ix,     iy + 1, (1.f - fx) * fy);
    TAP(ix + 1, iy,     fx * (1.f - fy));
    TAP(ix + 1, iy + 1, fx * fy);
#undef TAP
    out[idx] = v;
  }
}

// convex-combination 8x upsampling; mask[576][HW] already holds 0.25*(conv+b)
__global__ void k_upsample_flow(const float* __restrict__ flow,
                                const float* __restrict__ mask,
                                float* __restrict__ outz) {
  long total = 2L * HZ * WZ;
  GS_LOOP(idx, total) {
    int ch = (int)(idx / ((long)HZ * WZ));
    long r = idx - (long)ch * HZ * WZ;
    int uh = (int)(r / WZ), uw = (int)(r - (long)(r / WZ) * WZ);
    int h = uh >> 3, w = uw >> 3, i = uh & 7, j = uw & 7;
    float lg[9], mx = -1e30f;
#pragma unroll
    for (int k = 0; k < 9; ++k) {
      lg[k] = mask[(long)(((k * 8 + i) * 8) + j) * FHW + (long)h * FW + w];
      mx = fmaxf(mx, lg[k]);
    }
    float se = 0.f;
#pragma unroll
    for (int k = 0; k < 9; ++k) { lg[k] = __expf(lg[k] - mx); se += lg[k]; }
    float acc = 0.f;
#pragma unroll
    for (int k = 0; k < 9; ++k) {
      int hh = h + k / 3 - 1, ww = w + k % 3 - 1;
      float fv = (hh >= 0 && hh < FH && ww >= 0 && ww < FW)
                     ? 8.f * flow[(long)ch * FHW + (long)hh * FW + ww] : 0.f;
      acc += lg[k] * fv;
    }
    outz[idx] = acc / se;
  }
}

// ---------------------------------------------------------------------------
// Host orchestration
// ---------------------------------------------------------------------------
static inline dim3 gsz(long n) {
  long b = (n + 255) / 256;
  if (b < 1) b = 1;
  if (b > (1 << 20)) b = (1 << 20);
  return dim3((unsigned)b);
}
static inline int cdiv(int a, int b) { return (a + b - 1) / b; }

struct LayerW {
  const float* w;
  const float* b;
  int OC, K, Kt, Nt;
  half_t* bp;
};

extern "C" void kernel_launch(void* const* d_in, const int* in_sizes, int n_in,
                              void* d_out, int out_size, void* d_ws, size_t ws_size,
                              hipStream_t stream) {
  (void)in_sizes; (void)n_in; (void)out_size; (void)ws_size;
  hipStream_t st = stream;
  const float* im1 = (const float*)d_in[0];
  const float* im2 = (const float*)d_in[1];

  // ---- bump allocator over d_ws (needs ~220 MB) ----
  char* wsb = (char*)d_ws;
  size_t off = 0;
  auto alloc = [&](size_t bytes) -> void* {
    off = (off + 255) & ~(size_t)255;
    void* p = wsb + off;
    off += bytes;
    return p;
  };

  float* i1z   = (float*)alloc((size_t)3 * HZ * WZ * 4);
  float* i2z   = (float*)alloc((size_t)3 * HZ * WZ * 4);
  float* e1    = (float*)alloc((size_t)64 * H1 * W1 * 4);
  float* e2    = (float*)alloc((size_t)96 * H2 * W2 * 4);
  float* e3    = (float*)alloc((size_t)128 * FHW * 4);
  float* f1    = (float*)alloc((size_t)256 * FHW * 4);
  float* f2    = (float*)alloc((size_t)256 * FHW * 4);
  float* cnet  = (float*)alloc((size_t)256 * FHW * 4);
  float* net   = (float*)alloc((size_t)128 * FHW * 4);
  float* inp   = (float*)alloc((size_t)128 * FHW * 4);
  float* coords1 = (float*)alloc((size_t)2 * FHW * 4);
  float* pyr0  = (float*)alloc((size_t)FHW * FHW * 4);            // 65 MB -> L2
  float* pyr1  = (float*)alloc((size_t)FHW * L1H * L1W * 4);
  float* pyr2  = (float*)alloc((size_t)FHW * L2H * L2W * 4);
  float* pyr3  = (float*)alloc((size_t)FHW * L3H * L3W * 4);
  half_t* apack = (half_t*)alloc((size_t)20160 * 512 * 2);        // max Mt*Kt slabs
  half_t* acorr = (half_t*)alloc((size_t)252 * 8 * 512 * 2);
  half_t* bcorr = (half_t*)alloc((size_t)252 * 8 * 512 * 2);
  float* flow  = (float*)alloc((size_t)2 * FHW * 4);
  float* corrf = (float*)alloc((size_t)324 * FHW * 4);
  float* cor   = (float*)alloc((size_t)256 * FHW * 4);
  float* cf    = (float*)alloc((size_t)256 * FHW * 4);   // [mc2 out 192 ; mf2 out 64]
  float* flo   = (float*)alloc((size_t)128 * FHW * 4);
  float* hx    = (float*)alloc((size_t)384 * FHW * 4);   // [h ; inp ; mo(126) ; flow(2)]
  float* zb    = (float*)alloc((size_t)128 * FHW * 4);
  float* rb    = (float*)alloc((size_t)128 * FHW * 4);
  float* qb    = (float*)alloc((size_t)128 * FHW * 4);
  float* fhb   = (float*)alloc((size_t)256 * FHW * 4);
  float* dflow = (float*)alloc((size_t)2 * FHW * 4);
  float* um1o  = (float*)alloc((size_t)256 * FHW * 4);
  float* maskb = (float*)alloc((size_t)576 * FHW * 4);
  float* fzoom = (float*)alloc((size_t)2 * HZ * WZ * 4);
  float* fup   = (float*)alloc((size_t)2 * IHH * IWW * 4);

  // ---- pack all conv weights to WMMA B-fragment layout ----
  auto mkLayer = [&](int wi, int bi, int OC, int IC, int KH, int KW) -> LayerW {
    LayerW L;
    L.w = (const float*)d_in[wi];
    L.b = (const float*)d_in[bi];
    L.OC = OC;
    L.K = IC * KH * KW;
    L.Kt = cdiv(L.K, 32);
    L.Nt = cdiv(OC, 16);
    L.bp = (half_t*)alloc((size_t)L.Nt * L.Kt * 512 * 2);
    long total = (long)L.Nt * L.Kt * 512;
    k_pack_b<<<gsz(total), 256, 0, st>>>(L.w, L.bp, L.Nt, L.Kt, L.OC, L.K,
                                         1L, (long)L.K, 1.0f);
    return L;
  };
  // input order: im1, im2, then params in dict-insertion order (name_w, name_b)
  LayerW Lf1 = mkLayer(2, 3, 64, 3, 7, 7);
  LayerW Lf2 = mkLayer(4, 5, 96, 64, 3, 3);
  LayerW Lf3 = mkLayer(6, 7, 128, 96, 3, 3);
  LayerW Lf4 = mkLayer(8, 9, 256, 128, 1, 1);
  LayerW Lc1 = mkLayer(10, 11, 64, 3, 7, 7);
  LayerW Lc2 = mkLayer(12, 13, 96, 64, 3, 3);
  LayerW Lc3 = mkLayer(14, 15, 128, 96, 3, 3);
  LayerW Lc4 = mkLayer(16, 17, 256, 128, 1, 1);
  LayerW Lmc1 = mkLayer(18, 19, 256, 324, 1, 1);
  LayerW Lmc2 = mkLayer(20, 21, 192, 256, 3, 3);
  LayerW Lmf1 = mkLayer(22, 23, 128, 2, 7, 7);
  LayerW Lmf2 = mkLayer(24, 25, 64, 128, 3, 3);
  LayerW Lmo  = mkLayer(26, 27, 126, 256, 3, 3);
  LayerW Lgz1 = mkLayer(28, 29, 128, 384, 1, 5);
  LayerW Lgr1 = mkLayer(30, 31, 128, 384, 1, 5);
  LayerW Lgq1 = mkLayer(32, 33, 128, 384, 1, 5);
  LayerW Lgz2 = mkLayer(34, 35, 128, 384, 5, 1);
  LayerW Lgr2 = mkLayer(36, 37, 128, 384, 5, 1);
  LayerW Lgq2 = mkLayer(38, 39, 128, 384, 5, 1);
  LayerW Lfh1 = mkLayer(40, 41, 256, 128, 3, 3);
  LayerW Lfh2 = mkLayer(42, 43, 2, 256, 3, 3);
  LayerW Lum1 = mkLayer(44, 45, 256, 128, 3, 3);
  LayerW Lum2 = mkLayer(46, 47, 576, 256, 1, 1);

  auto gemm_go = [&](int act, const half_t* A, const half_t* B, const float* bias,
                     float* out, int M, int N, int Mt, int Nt, int Kt, float alpha) {
    long tiles = (long)((Mt + 1) / 2) * ((Nt + 1) / 2);
    long blk = (tiles + 7) / 8;
    if (blk > (1 << 20)) blk = (1 << 20);
    dim3 g((unsigned)blk), b(256);
    switch (act) {
      case 1: k_gemm<1><<<g, b, 0, st>>>(A, B, bias, out, M, N, Mt, Nt, Kt, alpha); break;
      case 2: k_gemm<2><<<g, b, 0, st>>>(A, B, bias, out, M, N, Mt, Nt, Kt, alpha); break;
      case 3: k_gemm<3><<<g, b, 0, st>>>(A, B, bias, out, M, N, Mt, Nt, Kt, alpha); break;
      default: k_gemm<0><<<g, b, 0, st>>>(A, B, bias, out, M, N, Mt, Nt, Kt, alpha); break;
    }
  };

  auto conv_apply = [&](const float* x, int IC, int IHs, int IWs, const LayerW& L,
                        int KH, int KW, int stride, int pH, int pW,
                        float* y, int act, float alpha) {
    int OH = (IHs + 2 * pH - KH) / stride + 1;
    int OW = (IWs + 2 * pW - KW) / stride + 1;
    int M = OH * OW, Mt = cdiv(M, 16);
    (void)IC;
    long at = (long)Mt * L.Kt * 512;
    k_pack_a_im2col<<<gsz(at), 256, 0, st>>>(x, apack, Mt, L.Kt, M, L.K, OW,
                                             KH, KW, IHs, IWs, stride, pH, pW);
    gemm_go(act, apack, L.bp, L.b, y, M, L.OC, Mt, L.Nt, L.Kt, alpha);
  };

  auto gru_pass = [&](const LayerW& Lz, const LayerW& Lr, const LayerW& Lq,
                      int KHg, int KWg, int pH, int pW) {
    hipMemcpyAsync(hx, net, (size_t)128 * FHW * 4, hipMemcpyDeviceToDevice, st);
    int M = (int)FHW, Mt = cdiv(M, 16);
    long at = (long)Mt * Lz.Kt * 512;
    k_pack_a_im2col<<<gsz(at), 256, 0, st>>>(hx, apack, Mt, Lz.Kt, M, Lz.K, FW,
                                             KHg, KWg, FH, FW, 1, pH, pW);
    gemm_go(2, apack, Lz.bp, Lz.b, zb, M, 128, Mt, Lz.Nt, Lz.Kt, 1.f);
    gemm_go(2, apack, Lr.bp, Lr.b, rb, M, 128, Mt, Lr.Nt, Lr.Kt, 1.f);
    k_mul<<<gsz(128 * FHW), 256, 0, st>>>(rb, net, hx, 128 * FHW);  // hx = [r*h ; x]
    k_pack_a_im2col<<<gsz(at), 256, 0, st>>>(hx, apack, Mt, Lq.Kt, M, Lq.K, FW,
                                             KHg, KWg, FH, FW, 1, pH, pW);
    gemm_go(3, apack, Lq.bp, Lq.b, qb, M, 128, Mt, Lq.Nt, Lq.Kt, 1.f);
    k_gru_combine<<<gsz(128 * FHW), 256, 0, st>>>(net, zb, qb, 128 * FHW);
  };

  // ---- 1) resize to 1.5x and normalize (2x-1) ----
  k_resize_bilinear<<<gsz(3L * HZ * WZ), 256, 0, st>>>(im1, i1z, 3, IHH, IWW, HZ, WZ, 2.f, -1.f);
  k_resize_bilinear<<<gsz(3L * HZ * WZ), 256, 0, st>>>(im2, i2z, 3, IHH, IWW, HZ, WZ, 2.f, -1.f);

  // ---- 2) encoders ----
  conv_apply(i1z, 3, HZ, WZ, Lf1, 7, 7, 2, 3, 3, e1, 1, 1.f);
  conv_apply(e1, 64, H1, W1, Lf2, 3, 3, 2, 1, 1, e2, 1, 1.f);
  conv_apply(e2, 96, H2, W2, Lf3, 3, 3, 2, 1, 1, e3, 1, 1.f);
  conv_apply(e3, 128, FH, FW, Lf4, 1, 1, 1, 0, 0, f1, 0, 1.f);
  conv_apply(i2z, 3, HZ, WZ, Lf1, 7, 7, 2, 3, 3, e1, 1, 1.f);
  conv_apply(e1, 64, H1, W1, Lf2, 3, 3, 2, 1, 1, e2, 1, 1.f);
  conv_apply(e2, 96, H2, W2, Lf3, 3, 3, 2, 1, 1, e3, 1, 1.f);
  conv_apply(e3, 128, FH, FW, Lf4, 1, 1, 1, 0, 0, f2, 0, 1.f);
  conv_apply(i1z, 3, HZ, WZ, Lc1, 7, 7, 2, 3, 3, e1, 1, 1.f);
  conv_apply(e1, 64, H1, W1, Lc2, 3, 3, 2, 1, 1, e2, 1, 1.f);
  conv_apply(e2, 96, H2, W2, Lc3, 3, 3, 2, 1, 1, e3, 1, 1.f);
  conv_apply(e3, 128, FH, FW, Lc4, 1, 1, 1, 0, 0, cnet, 0, 1.f);
  k_act_tanh<<<gsz(128 * FHW), 256, 0, st>>>(cnet, net, 128 * FHW);
  k_act_relu<<<gsz(128 * FHW), 256, 0, st>>>(cnet + 128 * FHW, inp, 128 * FHW);

  // ---- 3) correlation pyramid: pyr0[f1pos][f2pos] = f1.f2 / 16 via WMMA ----
  k_pack_a_mat<<<gsz(252L * 8 * 512), 256, 0, st>>>(f2, acorr, 252, 8, (int)FHW, 256,
                                                    FHW, 1.f / 16.f);
  k_pack_b<<<gsz(252L * 8 * 512), 256, 0, st>>>(f1, bcorr, 252, 8, (int)FHW, 256,
                                                FHW, 1L, 1.f);
  gemm_go(0, acorr, bcorr, nullptr, pyr0, (int)FHW, (int)FHW, 252, 252, 8, 1.f);
  k_avgpool<<<gsz(FHW * L1H * L1W), 256, 0, st>>>(pyr0, pyr1, (int)FHW, FH, FW, L1H, L1W, 2, 2, 0);
  k_avgpool<<<gsz(FHW * L2H * L2W), 256, 0, st>>>(pyr1, pyr2, (int)FHW, L1H, L1W, L2H, L2W, 2, 2, 0);
  k_avgpool<<<gsz(FHW * L3H * L3W), 256, 0, st>>>(pyr2, pyr3, (int)FHW, L2H, L2W, L3H, L3W, 2, 2, 0);

  // ---- 4) init recurrent state ----
  k_coords_init<<<gsz(2 * FHW), 256, 0, st>>>(coords1);
  hipMemcpyAsync(hx + (size_t)128 * FHW, inp, (size_t)128 * FHW * 4,
                 hipMemcpyDeviceToDevice, st);  // x-part [128:256) = inp, constant

  // ---- 5) 32 update iterations ----
  for (int it = 0; it < RAFT_ITERS; ++it) {
    k_flow_from_coords<<<gsz(2 * FHW), 256, 0, st>>>(coords1, flow);
    k_corr_lookup<<<gsz(324 * FHW), 256, 0, st>>>(pyr0, pyr1, pyr2, pyr3, coords1, corrf);
    conv_apply(corrf, 324, FH, FW, Lmc1, 1, 1, 1, 0, 0, cor, 1, 1.f);
    conv_apply(cor, 256, FH, FW, Lmc2, 3, 3, 1, 1, 1, cf, 1, 1.f);                  // 0..191
    conv_apply(flow, 2, FH, FW, Lmf1, 7, 7, 1, 3, 3, flo, 1, 1.f);
    conv_apply(flo, 128, FH, FW, Lmf2, 3, 3, 1, 1, 1, cf + (size_t)192 * FHW, 1, 1.f);
    conv_apply(cf, 256, FH, FW, Lmo, 3, 3, 1, 1, 1, hx + (size_t)256 * FHW, 1, 1.f); // mot
    hipMemcpyAsync(hx + (size_t)382 * FHW, flow, (size_t)2 * FHW * 4,
                   hipMemcpyDeviceToDevice, st);
    gru_pass(Lgz1, Lgr1, Lgq1, 1, 5, 0, 2);  // horizontal 1x5
    gru_pass(Lgz2, Lgr2, Lgq2, 5, 1, 2, 0);  // vertical 5x1
    conv_apply(net, 128, FH, FW, Lfh1, 3, 3, 1, 1, 1, fhb, 1, 1.f);
    conv_apply(fhb, 256, FH, FW, Lfh2, 3, 3, 1, 1, 1, dflow, 0, 1.f);
    k_add_inplace<<<gsz(2 * FHW), 256, 0, st>>>(coords1, dflow, 2 * FHW);
  }

  // ---- 6) convex upsample + final resize + 5x5 avg pool ----
  conv_apply(net, 128, FH, FW, Lum1, 3, 3, 1, 1, 1, um1o, 1, 1.f);
  conv_apply(um1o, 256, FH, FW, Lum2, 1, 1, 1, 0, 0, maskb, 0, 0.25f);
  k_flow_from_coords<<<gsz(2 * FHW), 256, 0, st>>>(coords1, flow);
  k_upsample_flow<<<gsz(2L * HZ * WZ), 256, 0, st>>>(flow, maskb, fzoom);
  k_resize_bilinear<<<gsz(2L * IHH * IWW), 256, 0, st>>>(fzoom, fup, 2, HZ, WZ,
                                                         IHH, IWW, 1.f / 1.5f, 0.f);
  k_avgpool<<<gsz(2L * IHH * IWW), 256, 0, st>>>(fup, (float*)d_out, 2, IHH, IWW,
                                                 IHH, IWW, 5, 1, 2);
}
